// GraphModel_13305808683297
// MI455X (gfx1250) — matrix operand order repaired and checked by reading, not compiled, not run on hardware
//
#include <hip/hip_runtime.h>
#include <hip/hip_bf16.h>

// ---------------------------------------------------------------------------
// GraphNet block, fused for MI455X (gfx1250, wave32, WMMA bf16 -> f32 acc).
//
//   N=8192 nodes, E=65536 edges, B=16 graphs, N_PER=512, E_PER=4096
//   node_enc 128->256->256+LN, edge_enc 128->256->256+LN, glob_enc 32->128->128+LN
//   edge_model 896->512->512+LN, node_model 896->512->512+LN,
//   glob_model 1152->256->256+LN
//
// d_in flattening assumption (setup_inputs() dict insertion order, params in
// insertion order, each MLP as Ws[0], Ws[1], bs[0], bs[1], g, b):
//   0 nodes  1 edges  2 glob  3 receivers  4 senders
//   5.. node_enc{W0,W1,b0,b1,g,b}  11.. edge_enc  17.. glob_enc
//   23.. edge_model  29.. node_model  35.. glob_model
//
// d_out = [nodes 8192x512 | edges 65536x512 | glob 16x256] f32.
//
// Encoder outputs are stored bf16 (they are only ever consumed as WMMA
// A-operands), halving intermediate-tensor HBM traffic; bf16 segments are
// staged into LDS with GLOBAL_LOAD_ASYNC_TO_LDS_B128 (ASYNCcnt).
// ---------------------------------------------------------------------------

#define USE_ASYNC_COPY 1

typedef __attribute__((ext_vector_type(16))) __bf16 v16bf;
typedef __attribute__((ext_vector_type(8)))  __bf16 v8bf;
typedef __attribute__((ext_vector_type(8)))  float  v8f;

#define LN_EPS 1e-5f

__device__ __forceinline__ unsigned short f2bf(float f) {
  union { float f; unsigned u; } x; x.f = f;
  unsigned u = x.u;
  u += 0x7FFFu + ((u >> 16) & 1u);   // round-to-nearest-even
  return (unsigned short)(u >> 16);
}

struct Seg {
  const void* src;   // rows [*, width]; f32 or bf16 per isbf
  const int*  idx;   // optional gather indices (row -> src row)
  int width;         // columns contributed to the concat (multiple of 8)
  int div;           // if idx==null and div>1: src row = row / div
  int isbf;          // 1: src is bf16 (raw copy), 0: f32 (convert)
};

struct MlpArgs {
  Seg segs[4];
  int nsegs;
  const unsigned short* W0;  // bf16 [N1][K1] row-major
  const unsigned short* W1;  // bf16 [N2][N1] row-major
  const float* b0;
  const float* b1;
  const float* gamma;
  const float* beta;
  void* out;                 // [M][N2]; f32 or bf16 per OUTBF
};

// Fused: concat-gather -> (x @ W0^T + b0) -> ReLU -> (@ W1^T + b1) -> LayerNorm
// One 256-thread block (8 waves) per 16-row tile. K1,N1,N2 multiples of 32;
// N1,N2 multiples of 128 (tiles split evenly over 8 waves).
template<int K1, int N1, int N2, int OUTBF>
__global__ __launch_bounds__(256)
void fused_mlp2_ln(MlpArgs a) {
  constexpr int K1P = K1 + 8;                  // bf16 row stride (16B aligned)
  constexpr int N1P = N1 + 8;
  constexpr int N2P = N2 + 4;                  // f32 row stride
  constexpr int XBYTES = 16 * K1P * 2;
  constexpr int YBYTES = 16 * N2P * 4;
  constexpr int R1 = (XBYTES > YBYTES) ? XBYTES : YBYTES;  // x/y overlap
  constexpr int HBYTES = 16 * N1P * 2;
  constexpr int T1 = N1 / 128;                 // 16-wide N tiles per wave
  constexpr int T2 = N2 / 128;

  __shared__ __align__(32) char smem[R1 + HBYTES];
  __shared__ int srcRow[4][16];

  unsigned short* xs = reinterpret_cast<unsigned short*>(smem);       // bf16 x
  float*          ys = reinterpret_cast<float*>(smem);                // f32 y
  unsigned short* hs = reinterpret_cast<unsigned short*>(smem + R1);  // bf16 h

  const int tid  = threadIdx.x;
  const int wave = tid >> 5;
  const int lane = tid & 31;
  const int hi   = lane >> 4;       // lane group 0/1
  const int lo16 = lane & 15;
  const int rowBase = blockIdx.x << 4;

  // Resolve gathered source rows for this 16-row tile.
  if (tid < 64) {
    int s = tid >> 4, r = tid & 15;
    int grow = rowBase + r;
    int sr = grow;
    if (s < a.nsegs) {
      const Seg& sg = a.segs[s];
      if (sg.idx)          sr = sg.idx[grow];
      else if (sg.div > 1) sr = grow / sg.div;
    }
    srcRow[s][r] = sr;
  }
  __syncthreads();

  // Stage x tile in 8-column (16B) chunks. Segment widths are multiples of 8
  // so a chunk never straddles segments. bf16 segments: async copy to LDS;
  // f32 segments: load + convert.
  constexpr int CPR = K1 / 8;            // chunks per row
  for (int e = tid; e < 16 * CPR; e += 256) {
    int r  = e / CPR;
    int c  = (e - r * CPR) * 8;
    int s = 0, base = 0;
    while (s + 1 < a.nsegs && c >= base + a.segs[s].width) {
      base += a.segs[s].width; ++s;
    }
    const Seg& sg = a.segs[s];
    unsigned short* dst = xs + r * K1P + c;
    size_t srcoff = (size_t)srcRow[s][r] * sg.width + (c - base);
    if (sg.isbf) {
      const unsigned short* gsrc = (const unsigned short*)sg.src + srcoff;
#if USE_ASYNC_COPY
      unsigned lds_off = (unsigned)(size_t)dst;     // low 32b = LDS byte offset
      unsigned long long gaddr = (unsigned long long)(size_t)gsrc;
      asm volatile("global_load_async_to_lds_b128 %0, %1, off"
                   :: "v"(lds_off), "v"(gaddr) : "memory");
#else
      *reinterpret_cast<v8bf*>(dst) = *reinterpret_cast<const v8bf*>(gsrc);
#endif
    } else {
      const float* f = (const float*)sg.src + srcoff;
      #pragma unroll
      for (int j = 0; j < 8; ++j) dst[j] = f2bf(f[j]);
    }
  }
#if USE_ASYNC_COPY
  asm volatile("s_wait_asynccnt 0x0" ::: "memory");
#endif
  __syncthreads();

  // ---- Layer 1: h = relu(x @ W0^T + b0) -------------------------------
  v8f acc[T1];
  #pragma unroll
  for (int t = 0; t < T1; ++t)
    #pragma unroll
    for (int i = 0; i < 8; ++i) acc[t][i] = 0.0f;

  for (int k0 = 0; k0 < K1; k0 += 32) {
    // A fragment (16x32 bf16): lane row = lane&15; K = 16*hi + 0..15
    // contiguous -> two 16B LDS loads.
    const unsigned short* xrow = xs + lo16 * K1P + k0 + 16 * hi;
    v8bf alo = *reinterpret_cast<const v8bf*>(xrow);
    v8bf ah2 = *reinterpret_cast<const v8bf*>(xrow + 8);
    v16bf afrag = __builtin_shufflevector(alo, ah2,
        0,1,2,3,4,5,6,7,8,9,10,11,12,13,14,15);
    #pragma unroll
    for (int t = 0; t < T1; ++t) {
      int n = ((wave + 8 * t) << 4) + lo16;
      // B fragment (32x16): lane col n, K = 16*hi + 0..15 contiguous.
      v16bf bfrag = *reinterpret_cast<const v16bf*>(
          a.W0 + (size_t)n * K1 + k0 + 16 * hi);
      acc[t] = __builtin_amdgcn_wmma_f32_16x16x32_bf16(
          false, afrag, false, bfrag, (short)0, acc[t], false, false);
    }
  }

  #pragma unroll
  for (int t = 0; t < T1; ++t) {
    int n = ((wave + 8 * t) << 4) + lo16;
    float bias = a.b0[n];
    #pragma unroll
    for (int i = 0; i < 8; ++i) {        // D element i -> row i + 8*hi
      float v = acc[t][i] + bias;
      hs[(i + 8 * hi) * N1P + n] = f2bf(v > 0.0f ? v : 0.0f);
    }
  }
  __syncthreads();

  // ---- Layer 2: y = h @ W1^T + b1 -------------------------------------
  v8f acc2[T2];
  #pragma unroll
  for (int t = 0; t < T2; ++t)
    #pragma unroll
    for (int i = 0; i < 8; ++i) acc2[t][i] = 0.0f;

  for (int k0 = 0; k0 < N1; k0 += 32) {
    const unsigned short* hrow = hs + lo16 * N1P + k0 + 16 * hi;
    v8bf alo = *reinterpret_cast<const v8bf*>(hrow);
    v8bf ah2 = *reinterpret_cast<const v8bf*>(hrow + 8);
    v16bf afrag = __builtin_shufflevector(alo, ah2,
        0,1,2,3,4,5,6,7,8,9,10,11,12,13,14,15);
    #pragma unroll
    for (int t = 0; t < T2; ++t) {
      int n = ((wave + 8 * t) << 4) + lo16;
      v16bf bfrag = *reinterpret_cast<const v16bf*>(
          a.W1 + (size_t)n * N1 + k0 + 16 * hi);
      acc2[t] = __builtin_amdgcn_wmma_f32_16x16x32_bf16(
          false, afrag, false, bfrag, (short)0, acc2[t], false, false);
    }
  }

  #pragma unroll
  for (int t = 0; t < T2; ++t) {
    int n = ((wave + 8 * t) << 4) + lo16;
    float bias = a.b1[n];
    #pragma unroll
    for (int i = 0; i < 8; ++i)
      ys[(i + 8 * hi) * N2P + n] = acc2[t][i] + bias;   // overlaps xs, safe
  }
  __syncthreads();

  // ---- LayerNorm per row (wave handles 2 rows; shuffle reduction) ------
  #pragma unroll
  for (int rr = 0; rr < 2; ++rr) {
    int r = wave * 2 + rr;
    float s = 0.0f, ss = 0.0f;
    for (int c = lane; c < N2; c += 32) {
      float v = ys[r * N2P + c];
      s += v; ss += v * v;
    }
    #pragma unroll
    for (int off = 16; off > 0; off >>= 1) {
      s  += __shfl_xor(s,  off, 32);
      ss += __shfl_xor(ss, off, 32);
    }
    float mu  = s * (1.0f / N2);
    float var = ss * (1.0f / N2) - mu * mu;     // population variance
    float inv = rsqrtf(var + LN_EPS);
    size_t orow = (size_t)(rowBase + r) * N2;
    for (int c = lane; c < N2; c += 32) {
      float v = (ys[r * N2P + c] - mu) * inv * a.gamma[c] + a.beta[c];
      if (OUTBF) ((unsigned short*)a.out)[orow + c] = f2bf(v);
      else       ((float*)a.out)[orow + c] = v;
    }
  }
}

// ---------------------------------------------------------------------------
__global__ void cvt_f32_bf16(const float* __restrict__ src,
                             unsigned short* __restrict__ dst, int n) {
  int i = blockIdx.x * blockDim.x + threadIdx.x;
  if (i < n) dst[i] = f2bf(src[i]);
}

__global__ void zero_f32(float* __restrict__ p, int n) {
  int i = blockIdx.x * blockDim.x + threadIdx.x;
  if (i < n) p[i] = 0.0f;
}

// segment_sum(edges_out, receivers) -> agg[N,512] via f32 global atomics.
__global__ void scatter_add_edges(const float* __restrict__ eout,
                                  const int* __restrict__ recv,
                                  float* __restrict__ agg, int E) {
  long long t = (long long)blockIdx.x * blockDim.x + threadIdx.x;
  long long total = (long long)E * 128;          // 512 cols / 4 per thread
  if (t >= total) return;
  int e  = (int)(t >> 7);
  int c4 = ((int)(t & 127)) << 2;
  const float4 v = *reinterpret_cast<const float4*>(eout + (size_t)e * 512 + c4);
  float* dst = agg + (size_t)recv[e] * 512 + c4;
  atomicAdd(dst + 0, v.x);
  atomicAdd(dst + 1, v.y);
  atomicAdd(dst + 2, v.z);
  atomicAdd(dst + 3, v.w);
}

// Per-graph sums over contiguous row ranges (deterministic).
__global__ void graph_sum(const float* __restrict__ src, float* __restrict__ dst,
                          int rowsPerGraph, int width) {
  int g = blockIdx.x;
  int c = blockIdx.y * blockDim.x + threadIdx.x;
  if (c >= width) return;
  const float* p = src + (size_t)g * rowsPerGraph * width + c;
  float s = 0.0f;
  for (int r = 0; r < rowsPerGraph; ++r) s += p[(size_t)r * width];
  dst[(size_t)g * width + c] = s;
}

// ---------------------------------------------------------------------------
extern "C" void kernel_launch(void* const* d_in, const int* in_sizes, int n_in,
                              void* d_out, int out_size, void* d_ws, size_t ws_size,
                              hipStream_t stream) {
  (void)in_sizes; (void)n_in; (void)out_size; (void)ws_size;

  constexpr int N = 8192, E = 65536, B = 16, NPER = 512, EPER = 4096;

  const float* nodes = (const float*)d_in[0];
  const float* edges = (const float*)d_in[1];
  const float* glob  = (const float*)d_in[2];
  const int*   recv  = (const int*)d_in[3];
  const int*   send  = (const int*)d_in[4];

  // per-MLP base indices: +0 W0, +1 W1, +2 b0, +3 b1, +4 g, +5 b
  constexpr int P_NE = 5, P_EE = 11, P_GE = 17, P_EM = 23, P_NM = 29, P_GM = 35;
  auto F = [&](int i) { return (const float*)d_in[i]; };

  // ---- workspace layout -------------------------------------------------
  char* ws = (char*)d_ws;
  size_t off = 0;
  auto alloc = [&](size_t bytes) -> void* {
    void* r = ws + off;
    off += (bytes + 255) & ~(size_t)255;
    return r;
  };

  struct WInfo { int pidx; int elems; };
  const WInfo winfo[12] = {
    {P_NE + 0, 256 * 128},  {P_NE + 1, 256 * 256},
    {P_EE + 0, 256 * 128},  {P_EE + 1, 256 * 256},
    {P_GE + 0, 128 * 32},   {P_GE + 1, 128 * 128},
    {P_EM + 0, 512 * 896},  {P_EM + 1, 512 * 512},
    {P_NM + 0, 512 * 896},  {P_NM + 1, 512 * 512},
    {P_GM + 0, 256 * 1152}, {P_GM + 1, 256 * 256},
  };
  unsigned short* wbf[12];
  for (int i = 0; i < 12; ++i)
    wbf[i] = (unsigned short*)alloc((size_t)winfo[i].elems * 2);

  unsigned short* encN = (unsigned short*)alloc((size_t)N * 256 * 2);  // bf16
  unsigned short* encE = (unsigned short*)alloc((size_t)E * 256 * 2);  // bf16
  unsigned short* encG = (unsigned short*)alloc((size_t)B * 128 * 2);  // bf16
  float* agg  = (float*)alloc((size_t)N * 512 * 4);
  float* nagg = (float*)alloc((size_t)B * 512 * 4);
  float* eagg = (float*)alloc((size_t)B * 512 * 4);

  float* out_nodes = (float*)d_out;
  float* out_edges = out_nodes + (size_t)N * 512;
  float* out_glob  = out_edges + (size_t)E * 512;

  // ---- weight conversion f32 -> bf16 (weights stay L2-resident) ---------
  for (int i = 0; i < 12; ++i) {
    int n = winfo[i].elems;
    cvt_f32_bf16<<<(n + 255) / 256, 256, 0, stream>>>(F(winfo[i].pidx), wbf[i], n);
  }

  // ---- encoders (bf16 outputs) ------------------------------------------
  {
    MlpArgs a{};
    a.segs[0] = {nodes, nullptr, 128, 0, 0};
    a.nsegs = 1; a.W0 = wbf[0]; a.W1 = wbf[1];
    a.b0 = F(P_NE + 2); a.b1 = F(P_NE + 3); a.gamma = F(P_NE + 4); a.beta = F(P_NE + 5);
    a.out = encN;
    fused_mlp2_ln<128, 256, 256, 1><<<N / 16, 256, 0, stream>>>(a);
  }
  {
    MlpArgs a{};
    a.segs[0] = {edges, nullptr, 128, 0, 0};
    a.nsegs = 1; a.W0 = wbf[2]; a.W1 = wbf[3];
    a.b0 = F(P_EE + 2); a.b1 = F(P_EE + 3); a.gamma = F(P_EE + 4); a.beta = F(P_EE + 5);
    a.out = encE;
    fused_mlp2_ln<128, 256, 256, 1><<<E / 16, 256, 0, stream>>>(a);
  }
  {
    MlpArgs a{};
    a.segs[0] = {glob, nullptr, 32, 0, 0};
    a.nsegs = 1; a.W0 = wbf[4]; a.W1 = wbf[5];
    a.b0 = F(P_GE + 2); a.b1 = F(P_GE + 3); a.gamma = F(P_GE + 4); a.beta = F(P_GE + 5);
    a.out = encG;
    fused_mlp2_ln<32, 128, 128, 1><<<B / 16, 256, 0, stream>>>(a);
  }

  // ---- edge model: concat[nodes[recv], nodes[send], edges, glob[gid]] ---
  // all segments bf16 -> async-to-LDS staging
  {
    MlpArgs a{};
    a.segs[0] = {encN, recv, 256, 0, 1};
    a.segs[1] = {encN, send, 256, 0, 1};
    a.segs[2] = {encE, nullptr, 256, 0, 1};
    a.segs[3] = {encG, nullptr, 128, EPER, 1};
    a.nsegs = 4; a.W0 = wbf[6]; a.W1 = wbf[7];
    a.b0 = F(P_EM + 2); a.b1 = F(P_EM + 3); a.gamma = F(P_EM + 4); a.beta = F(P_EM + 5);
    a.out = out_edges;
    fused_mlp2_ln<896, 512, 512, 0><<<E / 16, 256, 0, stream>>>(a);
  }

  // ---- scatter-sum edges -> nodes ---------------------------------------
  {
    int n = N * 512;
    zero_f32<<<(n + 255) / 256, 256, 0, stream>>>(agg, n);
    long long total = (long long)E * 128;
    scatter_add_edges<<<(unsigned)((total + 255) / 256), 256, 0, stream>>>(
        out_edges, recv, agg, E);
  }

  // ---- node model: concat[nodes, agg, glob[gid]] ------------------------
  {
    MlpArgs a{};
    a.segs[0] = {encN, nullptr, 256, 0, 1};
    a.segs[1] = {agg,  nullptr, 512, 0, 0};
    a.segs[2] = {encG, nullptr, 128, NPER, 1};
    a.nsegs = 3; a.W0 = wbf[8]; a.W1 = wbf[9];
    a.b0 = F(P_NM + 2); a.b1 = F(P_NM + 3); a.gamma = F(P_NM + 4); a.beta = F(P_NM + 5);
    a.out = out_nodes;
    fused_mlp2_ln<896, 512, 512, 0><<<N / 16, 256, 0, stream>>>(a);
  }

  // ---- per-graph sums ---------------------------------------------------
  {
    dim3 grid(B, 2);
    graph_sum<<<grid, 256, 0, stream>>>(out_nodes, nagg, NPER, 512);
    graph_sum<<<grid, 256, 0, stream>>>(out_edges, eagg, EPER, 512);
  }

  // ---- global model: concat[glob, nodes_agg, edges_agg] -----------------
  {
    MlpArgs a{};
    a.segs[0] = {encG, nullptr, 128, 0, 1};
    a.segs[1] = {nagg, nullptr, 512, 0, 0};
    a.segs[2] = {eagg, nullptr, 512, 0, 0};
    a.nsegs = 3; a.W0 = wbf[10]; a.W1 = wbf[11];
    a.b0 = F(P_GM + 2); a.b1 = F(P_GM + 3); a.gamma = F(P_GM + 4); a.beta = F(P_GM + 5);
    a.out = out_glob;
    fused_mlp2_ln<1152, 256, 256, 0><<<B / 16, 256, 0, stream>>>(a);
  }
}
